// Net_70025146794268
// MI455X (gfx1250) — compile-verified, matrix operand order checked by reading, not run on hardware
//
#include <hip/hip_runtime.h>
#include <hip/hip_bf16.h>

// ---------------------------------------------------------------------------
// PNA GNN for MI455X (gfx1250, wave32, WMMA).
// bf16 WMMA for all big matmuls; one graph per workgroup; node-feature tile
// staged into LDS by the Tensor Data Mover (TDM) when available; segment
// aggregation via LDS atomics; enc/pre and post/lin weights fused and packed
// device-side into WMMA B-fragment layout.
// ---------------------------------------------------------------------------

typedef __attribute__((ext_vector_type(16))) __bf16   v16bf;
typedef __attribute__((ext_vector_type(8)))  __bf16   v8bf;
typedef __attribute__((ext_vector_type(8)))  float    v8f;
typedef __attribute__((ext_vector_type(4)))  unsigned v4u;
typedef __attribute__((ext_vector_type(8)))  int      v8i;
typedef __attribute__((ext_vector_type(4)))  int      v4i;

#define Gc    1024
#define NPGc  126
#define EPGc  450
#define Nc    (Gc * NPGc)       // 129024
#define Ec    (Gc * EPGc)       // 460800
#define Fc    126
#define FPc   128
#define EPSc  1e-5f
#define SMEM_BYTES (32768 + 4 * 126 * 32 * 4)   // h tile + 4 stat planes = 97280

__device__ __forceinline__ v8f wmma_bf16(v16bf a, v16bf b, v8f c) {
    return __builtin_amdgcn_wmma_f32_16x16x32_bf16(false, a, false, b,
                                                   (short)0, c, false, false);
}
__device__ __forceinline__ v16bf cat16(v8bf lo, v8bf hi) {
    return __builtin_shufflevector(lo, hi, 0, 1, 2, 3, 4, 5, 6, 7,
                                   8, 9, 10, 11, 12, 13, 14, 15);
}
// monotone float <-> uint mapping so integer min/max == float min/max
__device__ __forceinline__ unsigned ordenc(float f) {
    unsigned b = __float_as_uint(f);
    return (b & 0x80000000u) ? ~b : (b | 0x80000000u);
}
__device__ __forceinline__ float orddec(unsigned u) {
    unsigned b = (u & 0x80000000u) ? (u & 0x7FFFFFFFu) : ~u;
    return __uint_as_float(b);
}
__device__ __forceinline__ float finfix(float v) {   // jnp.where(isfinite, v, 0)
    unsigned e = __float_as_uint(v) & 0x7F800000u;
    return (e == 0x7F800000u) ? 0.f : v;
}

// --------------------------- small prep kernels ----------------------------

__global__ void zero_f32_kernel(float* p, int n) {
    int i = blockIdx.x * 256 + threadIdx.x;
    if (i < n) p[i] = 0.f;
}

__global__ void avg_log_kernel(const int* dh, float* scal) {
    if (threadIdx.x == 0 && blockIdx.x == 0) {
        float s0 = 0.f, s1 = 0.f;
        for (int i = 0; i < 5; ++i) {
            float d = (float)dh[i];
            s0 += d;
            s1 += d * logf((float)i + 1.0f);
        }
        scal[0] = s1 / s0;
    }
}

__global__ void deg_kernel(const int* dst, float* cnt) {
    int e = blockIdx.x * 256 + threadIdx.x;
    if (e < Ec) atomicAdd(&cnt[dst[e]], 1.0f);
}

__global__ void scaler_kernel(const float* cnt, const float* scal,
                              float* inv, float* amp, float* att) {
    int n = blockIdx.x * 256 + threadIdx.x;
    if (n >= Nc) return;
    float avg_log = scal[0];
    float c  = cnt[n];
    float cm = fmaxf(c, 1.0f);
    float lg = logf(cm + 1.0f);
    inv[n] = 1.0f / cm;
    amp[n] = lg / avg_log;
    att[n] = avg_log / lg;
}

__global__ void embed_kernel(const float* x, const float* w, const float* b,
                             __bf16* h) {
    int id = blockIdx.x * 256 + threadIdx.x;   // over N*128
    int n = id >> 7, f = id & 127;
    float v = 0.f;
    if (f < Fc) {
        v = b[f];
        for (int k = 0; k < 5; ++k) v += x[n * 5 + k] * w[k * Fc + f];
    }
    h[id] = (__bf16)v;
}

__global__ void ea_kernel(const int* eattr, const float* tab, __bf16* ea) {
    long long id = (long long)blockIdx.x * 256 + threadIdx.x;  // over E*64
    if (id >= (long long)Ec * 64) return;
    int e = (int)(id >> 6), j = (int)(id & 63);
    float v = (j < 50) ? tab[eattr[e] * 50 + j] : 0.f;
    ea[id] = (__bf16)v;
}

// Pack fused message weight [h_dst(128)|h_src(128)|ea(64)] x 128 into
// B-fragment layout: [layer][kt(10)][nt(8)][lane(32)][16]  (bf16)
__global__ void pack_msg_kernel(const float* pre_w, const float* enc_w,
                                __bf16* Wmsg) {
    int id = blockIdx.x * 256 + threadIdx.x;
    const int per = 10 * 8 * 512;
    if (id >= 4 * per) return;
    int l = id / per;          int r  = id - l * per;
    int kt = r / 4096;         int r2 = r - kt * 4096;
    int nt = r2 / 512;         int r3 = r2 - nt * 512;
    int lane = r3 >> 4, j = r3 & 15, hi = lane >> 4;
    int n  = nt * 16 + (lane & 15);
    int kk = (j < 8) ? (8 * hi + j) : (16 + 8 * hi + (j - 8));
    int k  = kt * 32 + kk;
    float w = 0.f;
    if (n < Fc) {
        if (k < 128) {
            if (k < Fc) w = pre_w[(l * 378 + k) * Fc + n];              // h_dst
        } else if (k < 256) {
            int q = k - 128;
            if (q < Fc) w = pre_w[(l * 378 + 126 + q) * Fc + n];        // h_src
        } else {
            int q = k - 256;                                            // ea region
            if (q < 50)
                for (int c = 0; c < Fc; ++c)
                    w += enc_w[(l * 50 + q) * Fc + c] *
                         pre_w[(l * 378 + 252 + c) * Fc + n];
        }
    }
    Wmsg[id] = (__bf16)w;
}

__global__ void bias_msg_kernel(const float* pre_w, const float* pre_b,
                                const float* enc_b, float* bmsg) {
    int id = blockIdx.x * 256 + threadIdx.x;   // 4*128
    if (id >= 4 * 128) return;
    int l = id >> 7, n = id & 127;
    float b = 0.f;
    if (n < Fc) {
        b = pre_b[l * Fc + n];
        for (int c = 0; c < Fc; ++c)
            b += enc_b[l * Fc + c] * pre_w[(l * 378 + 252 + c) * Fc + n];
    }
    bmsg[id] = b;
}

// Pack Q = post_w_slice @ lin_w into B-fragment layout.
// which=0: Q1 (KT=20: kt 0..3 = h rows, kt 4..19 = s rows, feat-group permuted)
// which=1: Q2 (KT=16, s*amp rows); which=2: Q3 (KT=16, s*att rows)
__global__ void pack_upd_kernel(const float* post_w, const float* lin_w,
                                __bf16* Q, int which, int KT) {
    int id = blockIdx.x * 256 + threadIdx.x;
    int per = KT * 8 * 512;
    if (id >= 4 * per) return;
    int l = id / per;          int r  = id - l * per;
    int kt = r / 4096;         int r2 = r - kt * 4096;
    int nt = r2 / 512;         int r3 = r2 - nt * 512;
    int lane = r3 >> 4, j = r3 & 15, hi = lane >> 4;
    int n  = nt * 16 + (lane & 15);
    int kk = (j < 8) ? (8 * hi + j) : (16 + 8 * hi + (j - 8));
    int k  = kt * 32 + kk;
    int prow = -1;
    if (which == 0) {
        if (k < 128) { if (k < Fc) prow = k; }
        else {
            int ks = k - 128, fg = ks >> 7, rr = ks & 127;
            int stat = rr >> 5, feat = fg * 32 + (rr & 31);
            if (feat < Fc) prow = 126 + stat * 126 + feat;
        }
    } else {
        int ks = k, fg = ks >> 7, rr = ks & 127;
        int stat = rr >> 5, feat = fg * 32 + (rr & 31);
        if (feat < Fc) prow = 126 + which * 504 + stat * 126 + feat;
    }
    float w = 0.f;
    if (prow >= 0 && n < Fc)
        for (int c = 0; c < Fc; ++c)
            w += post_w[(l * 1638 + prow) * Fc + c] * lin_w[(l * Fc + c) * Fc + n];
    Q[id] = (__bf16)w;
}

__global__ void bias_upd_kernel(const float* post_b, const float* lin_w,
                                const float* lin_b, float* bupd) {
    int id = blockIdx.x * 256 + threadIdx.x;   // 4*128
    if (id >= 4 * 128) return;
    int l = id >> 7, n = id & 127;
    float b = 0.f;
    if (n < Fc) {
        b = lin_b[l * Fc + n];
        for (int c = 0; c < Fc; ++c)
            b += post_b[l * Fc + c] * lin_w[(l * Fc + c) * Fc + n];
    }
    bupd[id] = b;
}

// --------------------------- main PNA layer kernel -------------------------
// One graph per workgroup; 8 waves; wave w owns node tile [16w, 16w+16).
__global__ __launch_bounds__(256) void pna_layer_kernel(
    const __bf16* __restrict__ h,           // [N][128] bf16 (pad cols zero)
    const int*   __restrict__ srcIdx,
    const int*   __restrict__ dstIdx,
    const __bf16* __restrict__ ea,          // [E][64] bf16
    const float* __restrict__ invc,
    const float* __restrict__ amp,
    const float* __restrict__ att,
    const __bf16* __restrict__ Wmsg,        // [10][8][512]
    const float* __restrict__ bmsg,         // [128]
    const __bf16* __restrict__ Q1,          // [20][8][512]
    const __bf16* __restrict__ Q2,          // [16][8][512]
    const __bf16* __restrict__ Q3,          // [16][8][512]
    const float* __restrict__ bupd,         // [128]
    float* __restrict__ out,                // [N][128]
    float* __restrict__ bnsum, float* __restrict__ bnsq) {

    extern __shared__ char smem[];
    __bf16*   hs     = (__bf16*)smem;                     // [128][128] @ LDS 0
    float*    st_sum = (float*)(smem + 32768);            // [126][32]
    float*    st_sq  = st_sum + 126 * 32;
    unsigned* st_mn  = (unsigned*)(st_sq + 126 * 32);
    unsigned* st_mx  = st_mn + 126 * 32;

    const int g = blockIdx.x;
    const int tid = threadIdx.x;
    const int wave = tid >> 5, lane = tid & 31;
    const int hi = lane >> 4, lml = lane & 15;

    // pull weight fragments toward the caches early
    if (tid == 0) {
        __builtin_prefetch(Wmsg, 0, 1);
        __builtin_prefetch(Q1, 0, 1);
        __builtin_prefetch(Q2, 0, 1);
        __builtin_prefetch(Q3, 0, 1);
    }

    // ---- stage this graph's node-feature tile (126 rows x 256B) into LDS ----
#if __has_builtin(__builtin_amdgcn_tensor_load_to_lds) && \
    __has_builtin(__builtin_amdgcn_s_wait_tensorcnt)
    if (wave == 0) {
        unsigned long long ga =
            (unsigned long long)(const void*)(h + (size_t)g * NPGc * FPc);
        // D# group0: count=1 | lds_addr=0 | global_addr | type=2
        v4u g0 = { 1u, 0u, (unsigned)ga,
                   (unsigned)((ga >> 32) & 0x01FFFFFFu) | 0x80000000u };
        // D# group1: data_size=2B; tensor_dim0=128, tensor_dim1=126;
        // tile_dim0=128, tile_dim1=126; tensor_dim0_stride=128
        v8i g1 = { 0x00010000, (int)(128u << 16), (int)(126u << 16),
                   (int)(128u << 16), 126, 128, 0, 0 };
        v4i gz4 = { 0, 0, 0, 0 };
        v8i gz8 = { 0, 0, 0, 0, 0, 0, 0, 0 };
        __builtin_amdgcn_tensor_load_to_lds(g0, g1, gz4, gz4, gz8, 0);
        __builtin_amdgcn_s_wait_tensorcnt(0);
    }
#else
    {
        const __bf16* gsrc = h + (size_t)g * NPGc * FPc;
        for (int i = tid; i < (NPGc * FPc) / 8; i += 256)
            ((int4*)hs)[i] = ((const int4*)gsrc)[i];
    }
#endif
    // zero pad rows 126..127 of the LDS tile (disjoint from TDM target region)
    for (int i = tid; i < 2 * FPc; i += 256) hs[NPGc * FPc + i] = (__bf16)0.f;
    __syncthreads();

    // node-update accumulators (wave's 16-node tile x 128 out features, x3)
    v8f C1[8] = {}; v8f C2[8] = {}; v8f C3[8] = {};

    // ---- h @ Q0 (kt 0..3 of Q1) ----
    {
        int nodeA = 16 * wave + lml;
        const __bf16* hrow = hs + min(nodeA, NPGc - 1) * FPc;
        #pragma unroll
        for (int kt = 0; kt < 4; ++kt) {
            int kb = kt * 32 + 8 * hi;
            v16bf a = cat16(*(const v8bf*)(hrow + kb),
                            *(const v8bf*)(hrow + kb + 16));
            #pragma unroll
            for (int nt = 0; nt < 8; ++nt) {
                v16bf b = *(const v16bf*)(Q1 + (kt * 8 + nt) * 512 + lane * 16);
                C1[nt] = wmma_bf16(a, b, C1[nt]);
            }
        }
    }

    // ---- 4 feature-group passes (32 message feats each) ----
    for (int fg = 0; fg < 4; ++fg) {
        for (int i = tid; i < 126 * 32; i += 256) {
            st_sum[i] = 0.f; st_sq[i] = 0.f;
            st_mn[i] = 0xFFFFFFFFu; st_mx[i] = 0u;
        }
        __syncthreads();

        // messages for all edges, features [32fg, 32fg+32)
        for (int t = wave; t < 29; t += 8) {
            int eA = min(t * 16 + lml, EPGc - 1);
            size_t eg = (size_t)g * EPGc + eA;
            int dl = dstIdx[eg] - g * NPGc;
            int sl = srcIdx[eg] - g * NPGc;
            v8f Cm[2] = {};
            #pragma unroll
            for (int kt = 0; kt < 10; ++kt) {
                v16bf a;
                if (kt < 8) {
                    const __bf16* row = hs + (kt < 4 ? dl : sl) * FPc;
                    int kb = (kt & 3) * 32 + 8 * hi;
                    a = cat16(*(const v8bf*)(row + kb),
                              *(const v8bf*)(row + kb + 16));
                } else {
                    const __bf16* p = ea + eg * 64 + (kt - 8) * 32 + 8 * hi;
                    a = cat16(*(const v8bf*)p, *(const v8bf*)(p + 16));
                }
                #pragma unroll
                for (int j = 0; j < 2; ++j) {
                    int nt = 2 * fg + j;
                    v16bf b = *(const v16bf*)(Wmsg + (kt * 8 + nt) * 512 + lane * 16);
                    Cm[j] = wmma_bf16(a, b, Cm[j]);
                }
            }
            // aggregate into LDS (sum / sumsq / ordered-int min / max)
            #pragma unroll
            for (int j = 0; j < 2; ++j) {
                #pragma unroll
                for (int v = 0; v < 8; ++v) {
                    int em = t * 16 + v + 8 * hi;
                    if (em < EPGc) {
                        float m = Cm[j][v] + bmsg[(2 * fg + j) * 16 + lml];
                        int dle = dstIdx[(size_t)g * EPGc + em] - g * NPGc;
                        int idx = dle * 32 + j * 16 + lml;
                        atomicAdd(&st_sum[idx], m);
                        atomicAdd(&st_sq[idx], m * m);
                        unsigned u = ordenc(m);
                        atomicMin(&st_mn[idx], u);
                        atomicMax(&st_mx[idx], u);
                    }
                }
            }
        }
        __syncthreads();

        // update accumulation: s-chunk = [mean|min|max|std] (4 ktiles)
        {
            int nodeA = min(16 * wave + lml, NPGc - 1);
            float inv = invc[g * NPGc + nodeA];
            #pragma unroll
            for (int stat = 0; stat < 4; ++stat) {
                v16bf a;
                #pragma unroll
                for (int jj = 0; jj < 16; ++jj) {
                    int kk = (jj < 8) ? (8 * hi + jj) : (16 + 8 * hi + (jj - 8));
                    int sidx = nodeA * 32 + kk;
                    float val;
                    if (stat == 0)      val = st_sum[sidx] * inv;
                    else if (stat == 1) val = finfix(orddec(st_mn[sidx]));
                    else if (stat == 2) val = finfix(orddec(st_mx[sidx]));
                    else {
                        float mean = st_sum[sidx] * inv;
                        float v2 = st_sq[sidx] * inv - mean * mean;
                        val = sqrtf(fmaxf(v2, 0.f) + EPSc);
                    }
                    a[jj] = (__bf16)val;
                }
                int ktq1 = 4 + fg * 4 + stat;
                int ktq  = fg * 4 + stat;
                #pragma unroll
                for (int nt = 0; nt < 8; ++nt) {
                    v16bf b1 = *(const v16bf*)(Q1 + (ktq1 * 8 + nt) * 512 + lane * 16);
                    C1[nt] = wmma_bf16(a, b1, C1[nt]);
                    v16bf b2 = *(const v16bf*)(Q2 + (ktq * 8 + nt) * 512 + lane * 16);
                    C2[nt] = wmma_bf16(a, b2, C2[nt]);
                    v16bf b3 = *(const v16bf*)(Q3 + (ktq * 8 + nt) * 512 + lane * 16);
                    C3[nt] = wmma_bf16(a, b3, C3[nt]);
                }
            }
        }
        __syncthreads();
    }

    // ---- finalize: out = C1 + amp*C2 + att*C3 + b; per-block BN partials ----
    float* bsum = st_sum;           // reuse stats region
    float* bsq  = st_sum + 128;
    if (tid < 128) { bsum[tid] = 0.f; bsq[tid] = 0.f; }
    __syncthreads();
    #pragma unroll
    for (int nt = 0; nt < 8; ++nt) {
        #pragma unroll
        for (int v = 0; v < 8; ++v) {
            int node = 16 * wave + v + 8 * hi;
            int ncol = nt * 16 + lml;
            if (node < NPGc) {
                float a_ = amp[g * NPGc + node], t_ = att[g * NPGc + node];
                float val = C1[nt][v] + a_ * C2[nt][v] + t_ * C3[nt][v] + bupd[ncol];
                out[(size_t)(g * NPGc + node) * FPc + ncol] = val;
                if (ncol < Fc) {
                    atomicAdd(&bsum[ncol], val);
                    atomicAdd(&bsq[ncol], val * val);
                }
            }
        }
    }
    __syncthreads();
    if (tid < Fc) {
        atomicAdd(&bnsum[tid], bsum[tid]);
        atomicAdd(&bnsq[tid], bsq[tid]);
    }
}

// --------------------------- BN / pool / MLP tail --------------------------

__global__ void bn_reduce_kernel(const float* bnsum, const float* bnsq,
                                 float* mu, float* rstd) {
    int f = threadIdx.x;
    if (f < 128) {
        float m = (f < Fc) ? bnsum[f] / (float)Nc : 0.f;
        float v = (f < Fc) ? bnsq[f] / (float)Nc - m * m : 0.f;
        mu[f] = m;
        rstd[f] = rsqrtf(fmaxf(v, 0.f) + EPSc);
    }
}

__global__ void bn_apply_kernel(const float* outb, const float* mu,
                                const float* rstd, const float* g_,
                                const float* b_, __bf16* h) {
    int id = blockIdx.x * 256 + threadIdx.x;   // over N*128
    int f = id & 127;
    float v = 0.f;
    if (f < Fc)
        v = fmaxf(g_[f] * (outb[id] - mu[f]) * rstd[f] + b_[f], 0.f);
    h[id] = (__bf16)v;
}

__global__ void pool_kernel(const __bf16* h, float* pooled) {
    int id = blockIdx.x * 256 + threadIdx.x;   // over G*128
    int g = id >> 7, f = id & 127;
    if (g >= Gc || f >= Fc) return;
    float s = 0.f;
    for (int r = 0; r < NPGc; ++r)
        s += (float)h[(size_t)(g * NPGc + r) * FPc + f];
    pooled[g * Fc + f] = s;
}

__global__ __launch_bounds__(128) void mlp_kernel(
    const float* pooled,
    const float* w1, const float* b1, const float* w2, const float* b2,
    const float* w3, const float* b3, const float* w4, const float* b4,
    float* out) {
    __shared__ float in[126], z1[100], z2[50], z3[25];
    int g = blockIdx.x, t = threadIdx.x;
    if (t < 126) in[t] = pooled[g * 126 + t];
    __syncthreads();
    if (t < 100) {
        float s = b1[t];
        for (int k = 0; k < 126; ++k) s += in[k] * w1[k * 100 + t];
        z1[t] = fmaxf(s, 0.f);
    }
    __syncthreads();
    if (t < 50) {
        float s = b2[t];
        for (int k = 0; k < 100; ++k) s += z1[k] * w2[k * 50 + t];
        z2[t] = fmaxf(s, 0.f);
    }
    __syncthreads();
    if (t < 25) {
        float s = b3[t];
        for (int k = 0; k < 50; ++k) s += z2[k] * w3[k * 25 + t];
        z3[t] = fmaxf(s, 0.f);
    }
    __syncthreads();
    if (t == 0) {
        float s = b4[0];
        for (int k = 0; k < 25; ++k) s += z3[k] * w4[k];
        out[g] = s;
    }
}

// ------------------------------- launcher ----------------------------------

extern "C" void kernel_launch(void* const* d_in, const int* in_sizes, int n_in,
                              void* d_out, int out_size, void* d_ws, size_t ws_size,
                              hipStream_t stream) {
    (void)in_sizes; (void)n_in; (void)out_size; (void)ws_size;
    const float* x        = (const float*)d_in[0];
    const int*   ei       = (const int*)d_in[1];
    const int*   eattr    = (const int*)d_in[2];
    const int*   dh       = (const int*)d_in[4];
    const float* emb1_w   = (const float*)d_in[5];
    const float* emb1_b   = (const float*)d_in[6];
    const float* edge_tab = (const float*)d_in[7];
    const float* enc_w    = (const float*)d_in[8];
    const float* enc_b    = (const float*)d_in[9];
    const float* pre_w    = (const float*)d_in[10];
    const float* pre_b    = (const float*)d_in[11];
    const float* post_w   = (const float*)d_in[12];
    const float* post_b   = (const float*)d_in[13];
    const float* lin_w    = (const float*)d_in[14];
    const float* lin_b    = (const float*)d_in[15];
    const float* bn_g     = (const float*)d_in[16];
    const float* bn_b     = (const float*)d_in[17];
    const float* mw1 = (const float*)d_in[18]; const float* mb1 = (const float*)d_in[19];
    const float* mw2 = (const float*)d_in[20]; const float* mb2 = (const float*)d_in[21];
    const float* mw3 = (const float*)d_in[22]; const float* mb3 = (const float*)d_in[23];
    const float* mw4 = (const float*)d_in[24]; const float* mb4 = (const float*)d_in[25];

    const int* srcIdx = ei;
    const int* dstIdx = ei + Ec;

    char* base = (char*)d_ws;
    size_t off = 0;
    auto alloc = [&](size_t bytes) -> char* {
        char* p = base + off;
        off += (bytes + 255) & ~(size_t)255;
        return p;
    };
    __bf16* h      = (__bf16*)alloc((size_t)Nc * FPc * 2);
    float*  outb   = (float*)alloc((size_t)Nc * FPc * 4);
    __bf16* eab    = (__bf16*)alloc((size_t)Ec * 64 * 2);
    float*  cnt    = (float*)alloc((size_t)Nc * 4);
    float*  invc   = (float*)alloc((size_t)Nc * 4);
    float*  ampv   = (float*)alloc((size_t)Nc * 4);
    float*  attv   = (float*)alloc((size_t)Nc * 4);
    float*  scal   = (float*)alloc(64);
    float*  bnsum  = (float*)alloc(128 * 4);
    float*  bnsq   = (float*)alloc(128 * 4);
    float*  mu     = (float*)alloc(128 * 4);
    float*  rstd   = (float*)alloc(128 * 4);
    float*  pooled = (float*)alloc((size_t)Gc * Fc * 4);
    __bf16* Wmsg   = (__bf16*)alloc((size_t)4 * 10 * 8 * 512 * 2);
    float*  bmsg   = (float*)alloc(4 * 128 * 4);
    __bf16* Q1     = (__bf16*)alloc((size_t)4 * 20 * 8 * 512 * 2);
    __bf16* Q2     = (__bf16*)alloc((size_t)4 * 16 * 8 * 512 * 2);
    __bf16* Q3     = (__bf16*)alloc((size_t)4 * 16 * 8 * 512 * 2);
    float*  bupd   = (float*)alloc(4 * 128 * 4);

    // ---- prep ----
    avg_log_kernel<<<1, 1, 0, stream>>>(dh, scal);
    zero_f32_kernel<<<(Nc + 255) / 256, 256, 0, stream>>>(cnt, Nc);
    deg_kernel<<<(Ec + 255) / 256, 256, 0, stream>>>(dstIdx, cnt);
    scaler_kernel<<<(Nc + 255) / 256, 256, 0, stream>>>(cnt, scal, invc, ampv, attv);
    embed_kernel<<<(Nc * FPc) / 256, 256, 0, stream>>>(x, emb1_w, emb1_b, h);
    {
        long long tot = (long long)Ec * 64;
        ea_kernel<<<(unsigned)((tot + 255) / 256), 256, 0, stream>>>(eattr, edge_tab, eab);
    }
    pack_msg_kernel<<<(4 * 10 * 8 * 512 + 255) / 256, 256, 0, stream>>>(pre_w, enc_w, Wmsg);
    bias_msg_kernel<<<2, 256, 0, stream>>>(pre_w, pre_b, enc_b, bmsg);
    pack_upd_kernel<<<(4 * 20 * 8 * 512 + 255) / 256, 256, 0, stream>>>(post_w, lin_w, Q1, 0, 20);
    pack_upd_kernel<<<(4 * 16 * 8 * 512 + 255) / 256, 256, 0, stream>>>(post_w, lin_w, Q2, 1, 16);
    pack_upd_kernel<<<(4 * 16 * 8 * 512 + 255) / 256, 256, 0, stream>>>(post_w, lin_w, Q3, 2, 16);
    bias_upd_kernel<<<2, 256, 0, stream>>>(post_b, lin_w, lin_b, bupd);

    // ---- 4 PNA layers ----
    for (int l = 0; l < 4; ++l) {
        zero_f32_kernel<<<1, 256, 0, stream>>>(bnsum, 128);
        zero_f32_kernel<<<1, 256, 0, stream>>>(bnsq, 128);
        pna_layer_kernel<<<Gc, 256, SMEM_BYTES, stream>>>(
            h, srcIdx, dstIdx, eab, invc, ampv, attv,
            Wmsg + (size_t)l * 10 * 8 * 512, bmsg + l * 128,
            Q1 + (size_t)l * 20 * 8 * 512,
            Q2 + (size_t)l * 16 * 8 * 512,
            Q3 + (size_t)l * 16 * 8 * 512,
            bupd + l * 128, outb, bnsum, bnsq);
        bn_reduce_kernel<<<1, 128, 0, stream>>>(bnsum, bnsq, mu, rstd);
        bn_apply_kernel<<<(Nc * FPc) / 256, 256, 0, stream>>>(
            outb, mu, rstd, bn_g + l * Fc, bn_b + l * Fc, h);
    }

    // ---- pool + MLP head ----
    pool_kernel<<<(Gc * FPc) / 256, 256, 0, stream>>>(h, pooled);
    mlp_kernel<<<Gc, 128, 0, stream>>>(pooled, mw1, mb1, mw2, mb2, mw3, mb3,
                                       mw4, mb4, (float*)d_out);
}
